// Encoder_21492016349682
// MI455X (gfx1250) — matrix operand order repaired and checked by reading, not compiled, not run on hardware
//
#include <hip/hip_runtime.h>

typedef __attribute__((ext_vector_type(16))) _Float16     v16h;
typedef __attribute__((ext_vector_type(8)))  float        v8f;
typedef unsigned int u32;
typedef __attribute__((ext_vector_type(4)))  u32          v4u;
typedef __attribute__((ext_vector_type(8)))  int          v8i;
typedef __attribute__((ext_vector_type(4)))  int          v4i;

#define NUM_MODES 192
#define FEAT 384
#define LSEQ 1024
#define NROWS 32768          // 64*512
#define MTILE 128            // rows per block (8 waves x 16)
#define NWAVES 8

// Workspace layout (in _Float16 elements), all pre-swizzled B-fragment order
#define FSW_OFF   0
#define FSW_LEN   (LSEQ * FEAT)           // 393216
#define W1SW_OFF  (FSW_OFF + FSW_LEN)
#define W1SW_LEN  (384 * 256)
#define W2SW_OFF  (W1SW_OFF + W1SW_LEN)
#define W2SW_LEN  (256 * 128)
#define W3SW_OFF  (W2SW_OFF + W2SW_LEN)
#define W3SW_LEN  (128 * 32)

// LDS layout (bytes)
#define FEATF_B  (12 * NWAVES * 32 * 16 * 2)   // 98304: [kchunk][wave][lane][e]
#define H1F_B    (8  * NWAVES * 32 * 16 * 2)   // 65536
#define BBUF_B   (24 * 32 * 16 * 2)            // 24576 (max: F chunk 32x384)
#define XROWSTR  34                            // floats per x row in LDS (32 + 2 pad)
#define XBUF_B   (MTILE * XROWSTR * 4)         // 17408
#define OFF_FEATF 0
#define OFF_H1F   (OFF_FEATF + FEATF_B)
#define OFF_BBUF0 (OFF_H1F + H1F_B)
#define OFF_XBUF0 (OFF_BBUF0 + 2 * BBUF_B)
#define SMEM_B    (OFF_XBUF0 + 2 * XBUF_B)     // 247808 bytes (<320KB WGP LDS)

// x tile TDM padding: insert 2 dwords after every 32 dwords (interval code 4,
// amount code 1) -> LDS row stride 34 floats, 8B aligned, bank-spread.
#define XPADBITS ((1u << 20) | (4u << 22) | (1u << 25))

// ---------------------------------------------------------------------------
// Init kernels: pre-swizzle F (DFT matrix) and weights into B-fragment order:
//   idx = ((kchunk*ntiles + ntile)*32 + lane)*16 + e
//   holds B[k = kchunk*32 + (e>>3)*16 + (lane>>4)*8 + (e&7)][c = ntile*16 + (lane&15)]
// ---------------------------------------------------------------------------
__global__ void build_dft_sw(_Float16* __restrict__ Fsw) {
    int idx = blockIdx.x * 256 + threadIdx.x;
    if (idx >= FSW_LEN) return;
    int e    = idx & 15;
    int lane = (idx >> 4) & 31;
    int rest = idx >> 9;
    int nt   = rest % (FEAT / 16);
    int kc   = rest / (FEAT / 16);
    int t = kc * 32 + (e >> 3) * 16 + (lane >> 4) * 8 + (e & 7);
    int c = nt * 16 + (lane & 15);
    int m = c >> 1;
    int ph = (t * m) & (LSEQ - 1);
    float ang = 6.28318530717958647692f * (float)ph * (1.0f / (float)LSEQ);
    float v = (c & 1) ? -sinf(ang) : cosf(ang);
    Fsw[idx] = (_Float16)v;
}

__global__ void swizzle_weight(const float* __restrict__ W, _Float16* __restrict__ out,
                               int K, int N) {
    int idx = blockIdx.x * 256 + threadIdx.x;
    if (idx >= K * N) return;
    int e    = idx & 15;
    int lane = (idx >> 4) & 31;
    int rest = idx >> 9;
    int ntiles = N >> 4;
    int nt = rest % ntiles;
    int kc = rest / ntiles;
    int k = kc * 32 + (e >> 3) * 16 + (lane >> 4) * 8 + (e & 7);
    int c = nt * 16 + (lane & 15);
    out[idx] = (_Float16)W[(size_t)k * N + c];
}

// ---------------------------------------------------------------------------
// Tensor Data Mover: DMA a (rows x n8*8B) tile (row stride stride8*8B) from
// global to LDS. D# per CDNA5 ISA ch.8: group0 = {count/flags, lds_addr,
// global_addr lo, global_addr hi | type=2}; group1 packs data_size, dims,
// tile dims, strides, optional LDS padding. rows==1 -> 1D (tile_dim1=0).
// Unused descriptor groups (tensors <= 2D) are zero.
// ---------------------------------------------------------------------------
__device__ __forceinline__ void tdm_load(u32 lds_off, const void* g,
                                         u32 n8, u32 rows, u32 stride8,
                                         u32 padbits) {
    unsigned long long ga = (unsigned long long)(uintptr_t)g;
    v4u g0;
    g0[0] = 1u;                                              // count=1, user mode
    g0[1] = lds_off;                                         // LDS byte address
    g0[2] = (u32)ga;                                         // global_addr[31:0]
    g0[3] = (u32)((ga >> 32) & 0x01FFFFFFu) | (2u << 30);    // [56:32] | type=2
    v8i g1;
    g1[0] = (int)((3u << 16) | padbits);                     // data_size=8B, pad cfg
    g1[1] = (int)((n8 & 0xFFFFu) << 16);                     // tensor_dim0[15:0]
    g1[2] = (int)(((n8 >> 16) & 0xFFFFu) | ((rows & 0xFFFFu) << 16)); // td0 hi | td1 lo
    g1[3] = (int)(((rows >> 16) & 0xFFFFu) | ((n8 & 0xFFFFu) << 16)); // td1 hi | tile_dim0
    g1[4] = (int)(rows > 1 ? (rows & 0xFFFFu) : 0u);         // tile_dim1 (0 = 1D)
    g1[5] = (int)stride8;                                    // tensor_dim0_stride lo
    g1[6] = 0;
    g1[7] = 0;
    v4i z4 = {0, 0, 0, 0};
    v8i z8 = {0, 0, 0, 0, 0, 0, 0, 0};
    __builtin_amdgcn_tensor_load_to_lds(g0, g1, z4, z4, z8, 0);
}

__device__ __forceinline__ float leaky(float v) {
    return v > 0.0f ? v : 0.01f * v;
}

// Store a 16x16 f32 accumulator tile into an A-fragment LDS buffer
// (next layer's layout), applying bias (+ optional activation).
__device__ __forceinline__ void store_frag_a(_Float16* frag, int wave, int nt,
                                             int ln, int lh, const v8f& acc,
                                             float bias, bool act) {
    const int e  = (nt & 1) * 8 + (ln & 7);
    const int kc = nt >> 1;
#pragma unroll
    for (int v = 0; v < 8; ++v) {
        float r = acc[v] + bias;
        if (act) r = leaky(r);
        int lanep = ((ln >> 3) << 4) + v + 8 * lh;           // h'*16 + m
        frag[((kc * NWAVES + wave) * 32 + lanep) * 16 + e] = (_Float16)r;
    }
}

// Depth-1 software-pipelined WMMA sweep over NT 16-col tiles: issue the
// ds_load for B(nt+1) before wmma(nt) so waits carry one stage of slack.
#define WMMA_SWEEP(NT, ACC, AFRAG, BPTR)                                        \
    {                                                                           \
        v16h bcur = *(const v16h*)(BPTR);                                       \
        _Pragma("unroll")                                                       \
        for (int nt = 0; nt < (NT) - 1; ++nt) {                                 \
            v16h bnxt = *(const v16h*)((BPTR) + (nt + 1) * 512);                \
            ACC[nt] = __builtin_amdgcn_wmma_f32_16x16x32_f16(                   \
                false, AFRAG, false, bcur, (short)0, ACC[nt], false, false);    \
            bcur = bnxt;                                                        \
        }                                                                       \
        ACC[(NT) - 1] = __builtin_amdgcn_wmma_f32_16x16x32_f16(                 \
            false, AFRAG, false, bcur, (short)0, ACC[(NT) - 1], false, false);  \
    }

// ---------------------------------------------------------------------------
// Fused truncated-DFT GEMM + MLP(384->256->128->32), 128 rows per block.
// 256 threads = 8 waves; wave w owns rows [w*16, w*16+16).
// TDM double-buffers all global operands into LDS; inner loops are pure
// ds_load_b128 -> v_wmma_f32_16x16x32_f16.
// ---------------------------------------------------------------------------
__global__ void __launch_bounds__(256, 1)
fused_spectral_mlp(const float* __restrict__ x,
                   const _Float16* __restrict__ ws,
                   const float* __restrict__ b1,
                   const float* __restrict__ b2,
                   const float* __restrict__ b3,
                   float* __restrict__ out) {
    extern __shared__ char smem[];
    _Float16* featF = (_Float16*)(smem + OFF_FEATF);
    _Float16* h1F   = (_Float16*)(smem + OFF_H1F);
    _Float16* h2F   = featF;                                  // reuse

    const u32 sbase = (u32)(uintptr_t)smem;  // LDS byte base (low 32 bits)

    const _Float16* Fsw  = ws + FSW_OFF;
    const _Float16* W1sw = ws + W1SW_OFF;
    const _Float16* W2sw = ws + W2SW_OFF;
    const _Float16* W3sw = ws + W3SW_OFF;

    const int tid  = threadIdx.x;
    const int wave = tid >> 5;
    const int lane = tid & 31;
    const int ln   = lane & 15;
    const int lh   = lane >> 4;
    const int r0   = blockIdx.x * MTILE;
    const bool ldr = (wave == 0);

    const float* xblk = x + (size_t)r0 * LSEQ;
    const v8f vzero = {0.f, 0.f, 0.f, 0.f, 0.f, 0.f, 0.f, 0.f};

    // ---------------- Phase 1: feat = X @ F (K=1024, 32 chunks) ------------
    v8f acc1[24];
#pragma unroll
    for (int i = 0; i < 24; ++i) acc1[i] = vzero;

    if (ldr) {
        tdm_load(sbase + OFF_BBUF0, Fsw, 3072, 1, 3072, 0);            // 24KB F chunk
        tdm_load(sbase + OFF_XBUF0, xblk, 16, MTILE, 512, XPADBITS);   // 128x128B x tile
        __builtin_amdgcn_s_wait_tensorcnt(0);
    }
    __syncthreads();

    for (int kc5 = 0; kc5 < 32; ++kc5) {
        const int cur = kc5 & 1, nxt = cur ^ 1;
        if (ldr && kc5 + 1 < 32) {
            tdm_load(sbase + OFF_BBUF0 + nxt * BBUF_B,
                     Fsw + (size_t)(kc5 + 1) * (24 * 512), 3072, 1, 3072, 0);
            tdm_load(sbase + OFF_XBUF0 + nxt * XBUF_B,
                     xblk + (kc5 + 1) * 32, 16, MTILE, 512, XPADBITS);
        }
        // A fragment from LDS x tile (f32 -> f16 in registers)
        const float* xr = (const float*)(smem + OFF_XBUF0 + cur * XBUF_B)
                          + (wave * 16 + ln) * XROWSTR + lh * 8;
        v16h a;
#pragma unroll
        for (int pe = 0; pe < 8; ++pe) {
            float2 t = *(const float2*)(xr + ((pe >> 2) << 4) + ((pe & 3) << 1));
            a[2 * pe]     = (_Float16)t.x;
            a[2 * pe + 1] = (_Float16)t.y;
        }
        const _Float16* bc = (const _Float16*)(smem + OFF_BBUF0 + cur * BBUF_B)
                             + lane * 16;
        WMMA_SWEEP(24, acc1, a, bc);
        if (ldr) __builtin_amdgcn_s_wait_tensorcnt(0);
        __syncthreads();
    }

    if (ldr) tdm_load(sbase + OFF_BBUF0, W1sw, 2048, 1, 2048, 0);      // 16KB W1 chunk0
#pragma unroll
    for (int nt = 0; nt < 24; ++nt)
        store_frag_a(featF, wave, nt, ln, lh, acc1[nt], 0.0f, false);
    if (ldr) __builtin_amdgcn_s_wait_tensorcnt(0);
    __syncthreads();

    // ---------------- Phase 2: h1 = leaky(feat @ W1 + b1) (12 chunks) ------
    v8f acc2[16];
#pragma unroll
    for (int i = 0; i < 16; ++i) acc2[i] = vzero;

    for (int kc5 = 0; kc5 < 12; ++kc5) {
        const int cur = kc5 & 1, nxt = cur ^ 1;
        if (ldr && kc5 + 1 < 12)
            tdm_load(sbase + OFF_BBUF0 + nxt * BBUF_B,
                     W1sw + (size_t)(kc5 + 1) * (16 * 512), 2048, 1, 2048, 0);
        v16h a = *(const v16h*)(featF + ((kc5 * NWAVES + wave) * 32 + lane) * 16);
        const _Float16* bc = (const _Float16*)(smem + OFF_BBUF0 + cur * BBUF_B)
                             + lane * 16;
        WMMA_SWEEP(16, acc2, a, bc);
        if (ldr) __builtin_amdgcn_s_wait_tensorcnt(0);
        __syncthreads();
    }

    if (ldr) tdm_load(sbase + OFF_BBUF0, W2sw, 1024, 1, 1024, 0);      // 8KB W2 chunk0
#pragma unroll
    for (int nt = 0; nt < 16; ++nt)
        store_frag_a(h1F, wave, nt, ln, lh, acc2[nt], b1[nt * 16 + ln], true);
    if (ldr) __builtin_amdgcn_s_wait_tensorcnt(0);
    __syncthreads();

    // ---------------- Phase 3: h2 = leaky(h1 @ W2 + b2) (8 chunks) ---------
    v8f acc3[8];
#pragma unroll
    for (int i = 0; i < 8; ++i) acc3[i] = vzero;

    for (int kc5 = 0; kc5 < 8; ++kc5) {
        const int cur = kc5 & 1, nxt = cur ^ 1;
        if (ldr && kc5 + 1 < 8)
            tdm_load(sbase + OFF_BBUF0 + nxt * BBUF_B,
                     W2sw + (size_t)(kc5 + 1) * (8 * 512), 1024, 1, 1024, 0);
        v16h a = *(const v16h*)(h1F + ((kc5 * NWAVES + wave) * 32 + lane) * 16);
        const _Float16* bc = (const _Float16*)(smem + OFF_BBUF0 + cur * BBUF_B)
                             + lane * 16;
        WMMA_SWEEP(8, acc3, a, bc);
        if (ldr) __builtin_amdgcn_s_wait_tensorcnt(0);
        __syncthreads();
    }

    if (ldr) tdm_load(sbase + OFF_BBUF0, W3sw, 256, 1, 256, 0);        // 2KB W3 chunk0
#pragma unroll
    for (int nt = 0; nt < 8; ++nt)
        store_frag_a(h2F, wave, nt, ln, lh, acc3[nt], b2[nt * 16 + ln], true);
    if (ldr) __builtin_amdgcn_s_wait_tensorcnt(0);
    __syncthreads();

    // ---------------- Phase 4: out = h2 @ W3 + b3 (4 chunks) ---------------
    v8f acc4[2];
    acc4[0] = vzero; acc4[1] = vzero;

    for (int kc5 = 0; kc5 < 4; ++kc5) {
        const int cur = kc5 & 1, nxt = cur ^ 1;
        if (ldr && kc5 + 1 < 4)
            tdm_load(sbase + OFF_BBUF0 + nxt * BBUF_B,
                     W3sw + (size_t)(kc5 + 1) * (2 * 512), 256, 1, 256, 0);
        v16h a = *(const v16h*)(h2F + ((kc5 * NWAVES + wave) * 32 + lane) * 16);
        const _Float16* bc = (const _Float16*)(smem + OFF_BBUF0 + cur * BBUF_B)
                             + lane * 16;
        WMMA_SWEEP(2, acc4, a, bc);
        if (ldr) __builtin_amdgcn_s_wait_tensorcnt(0);
        __syncthreads();
    }

#pragma unroll
    for (int nt = 0; nt < 2; ++nt) {
        float bias = b3[nt * 16 + ln];
#pragma unroll
        for (int v = 0; v < 8; ++v) {
            int row = r0 + wave * 16 + v + 8 * lh;
            out[(size_t)row * 32 + nt * 16 + ln] = acc4[nt][v] + bias;
        }
    }
}

// ---------------------------------------------------------------------------
extern "C" void kernel_launch(void* const* d_in, const int* in_sizes, int n_in,
                              void* d_out, int out_size, void* d_ws, size_t ws_size,
                              hipStream_t stream) {
    const float* x  = (const float*)d_in[0];
    const float* W1 = (const float*)d_in[1];
    const float* b1 = (const float*)d_in[2];
    const float* W2 = (const float*)d_in[3];
    const float* b2 = (const float*)d_in[4];
    const float* W3 = (const float*)d_in[5];
    const float* b3 = (const float*)d_in[6];
    float* out = (float*)d_out;

    _Float16* ws = (_Float16*)d_ws;   // ~1.03 MB used

    build_dft_sw<<<(FSW_LEN + 255) / 256, 256, 0, stream>>>(ws + FSW_OFF);
    swizzle_weight<<<(W1SW_LEN + 255) / 256, 256, 0, stream>>>(W1, ws + W1SW_OFF, 384, 256);
    swizzle_weight<<<(W2SW_LEN + 255) / 256, 256, 0, stream>>>(W2, ws + W2SW_OFF, 256, 128);
    swizzle_weight<<<(W3SW_LEN + 255) / 256, 256, 0, stream>>>(W3, ws + W3SW_OFF, 128, 32);

    fused_spectral_mlp<<<NROWS / MTILE, 256, SMEM_B, stream>>>(
        x, ws, b1, b2, b3, out);
}
